// LSTM_32332513804762
// MI455X (gfx1250) — compile-verified, matrix operand order checked by reading, not run on hardware
//
#include <hip/hip_runtime.h>
#include <stdint.h>

typedef __attribute__((ext_vector_type(16))) __bf16 v16bf;
typedef __attribute__((ext_vector_type(8)))  float  v8f;
typedef int b128_t __attribute__((vector_size(16)));

union Frag16 { v16bf v; uint4 q[2]; };

#if __has_builtin(__builtin_amdgcn_global_load_async_to_lds_b128) && \
    __has_builtin(__builtin_amdgcn_s_wait_asynccnt)
#define LSTM_ASYNC 1
#else
#define LSTM_ASYNC 0
#endif

__device__ __forceinline__ unsigned short f2bf(float f) {
  union { float f; unsigned int u; } c; c.f = f;
  unsigned int u = c.u;
  unsigned int r = u + 0x7FFFu + ((u >> 16) & 1u);   // round-to-nearest-even
  return (unsigned short)(r >> 16);
}

__device__ __forceinline__ float sigmoidf_(float x) { return 1.0f / (1.0f + __expf(-x)); }

#if LSTM_ASYNC
__device__ __forceinline__ void async_cp16(const void* g, void* l) {
  __builtin_amdgcn_global_load_async_to_lds_b128(
      (__attribute__((address_space(1))) b128_t*)g,
      (__attribute__((address_space(3))) b128_t*)l, 0, 0);
}
#endif

// ---- Prep: Ws fp32 [L][K=2048][N=4096] -> bf16 transposed WT [L][N=4096][K=2048]
__global__ void wconv_kernel(const float* __restrict__ Ws, unsigned short* __restrict__ WT) {
  size_t idx = (size_t)blockIdx.x * blockDim.x + threadIdx.x;      // 2*4096*2048 threads
  size_t per = (size_t)4096 * 2048;
  size_t l = idx / per, rem = idx - l * per;
  size_t n = rem / 2048, k = rem - n * 2048;
  WT[idx] = f2bf(Ws[l * per + k * 4096 + n]);
}

// ---- Prep: x fp32 [B=64][T=256][D=1024] -> bf16 time-major xtm [T][B][D]
__global__ void xconv_kernel(const float* __restrict__ x, unsigned short* __restrict__ xtm) {
  size_t idx = (size_t)blockIdx.x * blockDim.x + threadIdx.x;      // 256*64*1024 threads
  size_t d = idx & 1023;
  size_t b = (idx >> 10) & 63;
  size_t t = idx >> 16;
  xtm[idx] = f2bf(x[(((b << 8) + t) << 10) + d]);
}

// ---- Prep: h0 fp32 [2][64][1024] -> bf16 h state + f32 c state
__global__ void init_kernel(const float* __restrict__ h0,
                            unsigned short* __restrict__ hA, float* __restrict__ c) {
  size_t idx = (size_t)blockIdx.x * blockDim.x + threadIdx.x;      // 2*65536 threads
  float v = h0[idx];
  hA[idx] = f2bf(v);
  c[idx]  = v;
}

// ---- One LSTM timestep: z = [X|Hin] @ W^T(+bias), gates, state update. Fused.
// Grid: 64 blocks (16 h-cols each, all 4 gates), 256 threads = 8 waves.
// wave = gate*2 + mhalf; each wave: 2 M-tiles (16x16) x 1 N-tile, K=2048 in 64 steps of 32.
// A chunks staged via async-to-LDS DMA, 4-deep circular buffer, prefetch distance 3.
__global__ __launch_bounds__(256) void lstm_step_kernel(
    const unsigned short* __restrict__ X,     // [64][1024] bf16
    const unsigned short* __restrict__ Hin,   // [64][1024] bf16
    const unsigned short* __restrict__ WT,    // [4096][2048] bf16 (n-major)
    const float* __restrict__ bias,           // [4096]
    float* __restrict__ C,                    // [64][1024] f32 (in/out)
    unsigned short* __restrict__ Hout,        // [64][1024] bf16 (next-step state)
    unsigned short* __restrict__ Hseq,        // [64][1024] bf16 (sequence out)
    float* __restrict__ Hf32)                 // [64][1024] f32  (FC head input)
{
#if LSTM_ASYNC
  __shared__ __align__(16) unsigned short lA[4][64 * 32];  // 4 x 4 KB A chunks
#else
  __shared__ __align__(16) unsigned short lA[1][64 * 32];
#endif
  __shared__ float zbuf[4][64][16];                        // per-gate z tile exchange (16 KB)

  const int tid  = threadIdx.x;
  const int lane = tid & 31;
  const int wave = tid >> 5;        // 0..7
  const int gate = wave >> 1;       // 0..3  (f,i,g,o)
  const int mh   = wave & 1;        // M half: tiles {0,1} or {2,3}
  const int hb   = blockIdx.x;      // h-column tile 0..63
  const int col  = lane & 15;
  const int hi   = lane >> 4;       // lane group 0/1
  const int koff = hi << 3;         // K sub-offset 0/8 (ISA 16-bit A/B lane layout)

  const unsigned short* brow = WT + (size_t)(gate * 1024 + hb * 16 + col) * 2048;

  v8f acc[2] = {};

  const int sr = tid >> 2;          // staging row 0..63
  const int sq = (tid & 3) << 3;    // staging k-offset {0,8,16,24}

  auto srcOf = [&](int kb, const unsigned short*& src, int& kloc) {
    if (kb < 32) { src = X;   kloc = kb << 5; }
    else         { src = Hin; kloc = (kb << 5) - 1024; }
  };

  auto loadB = [&](int kb) {
    Frag16 b;
    const unsigned short* p = brow + (kb << 5) + koff;
    b.q[0] = *(const uint4*)(p);
    b.q[1] = *(const uint4*)(p + 16);
    return b;
  };

  auto compute = [&](int kb, const Frag16& bF) {
#if LSTM_ASYNC
    const unsigned short* base = lA[kb & 3];
#else
    const unsigned short* base = lA[0]; (void)kb;
#endif
#pragma unroll
    for (int mt = 0; mt < 2; ++mt) {
      Frag16 aF;
      const unsigned short* p = &base[((mh * 2 + mt) * 16 + col) * 32 + koff];
      aF.q[0] = *(const uint4*)(p);
      aF.q[1] = *(const uint4*)(p + 16);
      acc[mt] = __builtin_amdgcn_wmma_f32_16x16x32_bf16(
          false, aF.v, false, bF.v, (short)0, acc[mt], false, false);
    }
  };

#if LSTM_ASYNC
  auto issue = [&](int kb) {
    const unsigned short* src; int kloc; srcOf(kb, src, kloc);
    async_cp16(&src[(size_t)sr * 1024 + kloc + sq], &lA[kb & 3][sr * 32 + sq]);
  };

  issue(0); issue(1); issue(2);
  Frag16 bF = loadB(0);
  for (int kb = 0; kb < 61; ++kb) {
    __builtin_amdgcn_s_wait_asynccnt(2);   // chunk kb landed (async done in order)
    __syncthreads();                       // visible to all waves; buf (kb-1)&3 free
    issue(kb + 3);
    Frag16 bN = loadB(kb + 1);             // B pipelined one iter ahead
    compute(kb, bF);
    bF = bN;
  }
  { __builtin_amdgcn_s_wait_asynccnt(2); __syncthreads(); Frag16 bN = loadB(62); compute(61, bF); bF = bN; }
  { __builtin_amdgcn_s_wait_asynccnt(1); __syncthreads(); Frag16 bN = loadB(63); compute(62, bF); bF = bN; }
  { __builtin_amdgcn_s_wait_asynccnt(0); __syncthreads(); compute(63, bF); }
#else
  for (int kb = 0; kb < 64; ++kb) {
    const unsigned short* src; int kloc; srcOf(kb, src, kloc);
    __syncthreads();
    *(uint4*)&lA[0][sr * 32 + sq] = *(const uint4*)&src[(size_t)sr * 1024 + kloc + sq];
    __syncthreads();
    Frag16 bF = loadB(kb);
    compute(kb, bF);
  }
#endif

  // Scatter z tiles to LDS: acc element j -> row m = tile*16 + j + hi*8, col = lane&15
#pragma unroll
  for (int mt = 0; mt < 2; ++mt)
#pragma unroll
    for (int j = 0; j < 8; ++j)
      zbuf[gate][(mh * 2 + mt) * 16 + j + (hi << 3)][col] = acc[mt][j];

  __syncthreads();

  // Fused gate math: 1024 (b,hcol) elements per block, 4 per thread
  const int hc0 = hb * 16;
  for (int e = tid; e < 1024; e += 256) {
    int b = e >> 4, cc = e & 15, hc = hc0 + cc;
    float f = zbuf[0][b][cc] + bias[hc];
    float i = zbuf[1][b][cc] + bias[1024 + hc];
    float g = zbuf[2][b][cc] + bias[2048 + hc];
    float o = zbuf[3][b][cc] + bias[3072 + hc];
    size_t off = ((size_t)b << 10) + hc;
    float cn = sigmoidf_(f) * C[off] + sigmoidf_(i) * tanhf(g);
    float hn = sigmoidf_(o) * tanhf(cn);
    C[off] = cn;
    unsigned short hb16 = f2bf(hn);
    Hout[off] = hb16;
    Hseq[off] = hb16;
    Hf32[off] = hn;
  }
}

// ---- FC head: out[64][1024] = Hlast @ Wfc + bfc (0.13 GFLOP, plain f32)
__global__ void fc_kernel(const float* __restrict__ H, const float* __restrict__ Wfc,
                          const float* __restrict__ bfc, float* __restrict__ out) {
  int gid = blockIdx.x * blockDim.x + threadIdx.x;   // 65536
  int b = gid >> 10, o = gid & 1023;
  float s = bfc[o];
  const float* hrow = H + ((size_t)b << 10);
  for (int h = 0; h < 1024; ++h) s += hrow[h] * Wfc[((size_t)h << 10) + o];
  out[gid] = s;
}

extern "C" void kernel_launch(void* const* d_in, const int* in_sizes, int n_in,
                              void* d_out, int out_size, void* d_ws, size_t ws_size,
                              hipStream_t stream) {
  const float* x   = (const float*)d_in[0];
  const float* h0  = (const float*)d_in[1];
  const float* Ws  = (const float*)d_in[2];
  const float* bs  = (const float*)d_in[3];
  const float* Wfc = (const float*)d_in[4];
  const float* bfc = (const float*)d_in[5];
  float* out = (float*)d_out;

  char* ws = (char*)d_ws;
  unsigned short* WT   = (unsigned short*)(ws);               // 2*4096*2048 bf16 = 32 MB
  unsigned short* xtm  = (unsigned short*)(ws + 33554432ull); // 256*64*1024 bf16 = 32 MB
  unsigned short* seq1 = (unsigned short*)(ws + 67108864ull); // 256*64*1024 bf16 = 32 MB
  unsigned short* hA   = (unsigned short*)(ws + 100663296ull);// [2][65536] bf16
  unsigned short* hB   = (unsigned short*)(ws + 100925440ull);// [2][65536] bf16
  float*          cst  = (float*)(ws + 101187584ull);         // [2][65536] f32
  float*          hlast= (float*)(ws + 101711872ull);         // [65536] f32

  wconv_kernel<<<65536, 256, 0, stream>>>(Ws, WT);
  xconv_kernel<<<65536, 256, 0, stream>>>(x, xtm);
  init_kernel<<<512, 256, 0, stream>>>(h0, hA, cst);

  // Layer 0: reads xtm[t], writes seq1[t]; h double-buffered across steps.
  for (int t = 0; t < 256; ++t) {
    unsigned short* hin  = (t & 1) ? hB : hA;
    unsigned short* hout = (t & 1) ? hA : hB;
    lstm_step_kernel<<<64, 256, 0, stream>>>(
        xtm + (size_t)t * 65536, hin, WT, bs, cst, hout,
        seq1 + (size_t)t * 65536, hlast);
  }
  // Layer 1: reads seq1[t]; final h (f32) feeds FC.
  for (int t = 0; t < 256; ++t) {
    unsigned short* hin  = ((t & 1) ? hB : hA) + 65536;
    unsigned short* hout = ((t & 1) ? hA : hB) + 65536;
    lstm_step_kernel<<<64, 256, 0, stream>>>(
        seq1 + (size_t)t * 65536, hin, WT + (size_t)4096 * 2048, bs + 4096,
        cst + 65536, hout, hout, hlast);
  }
  fc_kernel<<<256, 256, 0, stream>>>(hlast, Wfc, bfc, out);
}
